// MultiheadAttention_601295421902
// MI455X (gfx1250) — matrix operand order repaired and checked by reading, not compiled, not run on hardware
//
#include <hip/hip_runtime.h>

// ---------------------------------------------------------------------------
// MHA forward for gfx1250 (CDNA5, wave32) using v_wmma_f32_16x16x32_bf16.
// T=S=2048, B=2, E=1024, H=16, D=64.
// ---------------------------------------------------------------------------

typedef __attribute__((ext_vector_type(16))) __bf16 v16bf;
typedef __attribute__((ext_vector_type(8)))  __bf16 v8bf;
typedef __attribute__((ext_vector_type(8)))  float  v8f;

#define T_DIM 2048
#define B_DIM 2
#define E_DIM 1024
#define H_DIM 16
#define D_DIM 64
#define S_DIM 2048

// ---------------------------------------------------------------------------
// Prep: fuse positional adds, downcast to bf16, lay out operands so every
// WMMA fragment load is a contiguous 16B/32B per-lane load.
//   qb : [B][H][T][D]   (A-operand rows of Q·K^T)
//   kt : [B][H][D][S]   (pre-transposed -> B-operand rows of Q·K^T)
//   vb : [B][H][S][D]   (B-operand rows of P·V)
// ---------------------------------------------------------------------------
__global__ void prep_qkv(const float* __restrict__ q,  const float* __restrict__ qp,
                         const float* __restrict__ k,  const float* __restrict__ kp,
                         const float* __restrict__ v,
                         __bf16* __restrict__ qb, __bf16* __restrict__ kt,
                         __bf16* __restrict__ vb) {
  int idx = blockIdx.x * blockDim.x + threadIdx.x;      // [b][h][t][d]
  if (idx >= B_DIM * H_DIM * T_DIM * D_DIM) return;
  int d = idx & (D_DIM - 1);
  int t = (idx >> 6) & (T_DIM - 1);
  int h = (idx >> 17) & (H_DIM - 1);
  int b = idx >> 21;
  int e = h * D_DIM + d;
  size_t src = (size_t)(t * B_DIM + b) * E_DIM + e;
  qb[idx] = (__bf16)(q[src] + qp[src]);
  vb[idx] = (__bf16)v[src];
  kt[((size_t)(b * H_DIM + h) * D_DIM + d) * S_DIM + t] = (__bf16)(k[src] + kp[src]);
}

// Wt[k][n] = W[n][k]  (bf16) -> B-operand rows of the fused-heads GEMM
__global__ void prep_w(const float* __restrict__ W, __bf16* __restrict__ Wt) {
  int idx = blockIdx.x * blockDim.x + threadIdx.x;
  if (idx >= E_DIM * E_DIM) return;
  int n  = idx & (E_DIM - 1);
  int kk = idx >> 10;
  Wt[idx] = (__bf16)W[(size_t)n * E_DIM + kk];
}

// ---------------------------------------------------------------------------
// Attention: one 256-thread block per (b, 16-row query tile).  Loops all 16
// heads so the head-averaged attention accumulates in LDS (no atomics, and
// the 1 GB worth of per-head probabilities never touches HBM).
// Dynamic LDS (exactly 320 KB):
//   sc [16][2048] f32   raw scores -> exp values          (128 KB)
//   avg[16][2048] f32   running head-average               (128 KB)
//   pb [16][2048] bf16  normalized P, packed for A-frags   ( 64 KB)
// Softmax reductions use half-wave __shfl_xor (each row == one half-wave),
// so no LDS scratch and no barriers inside the softmax pass.
// ---------------------------------------------------------------------------
__global__ __launch_bounds__(256)
void attn_kernel(const __bf16* __restrict__ qb, const __bf16* __restrict__ kt,
                 const __bf16* __restrict__ vb, __bf16* __restrict__ ctx,
                 float* __restrict__ attn_out) {
  extern __shared__ float lds[];
  float*  sc  = lds;                         // 16*2048 f32
  float*  avg = sc + 16 * S_DIM;             // 16*2048 f32
  __bf16* pb  = (__bf16*)(avg + 16 * S_DIM); // 16*2048 bf16

  const int tid  = threadIdx.x;
  const int lane = tid & 31;            // wave32
  const int wave = tid >> 5;            // 8 waves
  const int hi8  = (lane >> 4) * 8;     // 16-bit A-operand lane split
  const int m16  = lane & 15;

  const int t0 = blockIdx.x * 16;
  const int b  = blockIdx.y;

  for (int i = tid; i < 16 * S_DIM; i += 256) avg[i] = 0.0f;
  __syncthreads();

  const float scale = 0.125f;           // 1/sqrt(D)

  for (int h = 0; h < H_DIM; ++h) {
    const int bh = b * H_DIM + h;

    // ---- Pass 1: scores = Q K^T (8 waves, 16-wide S strips) ----
    // A fragment (16x32 bf16): M = lane&15, K = 16*(e>>3) + 8*(lane>>4) + (e&7)
    const __bf16* qrow = qb + ((size_t)bh * T_DIM + (t0 + m16)) * D_DIM;
    v16bf a0, a1;
    {
      v8bf l0 = *(const v8bf*)(qrow + hi8);
      v8bf h0 = *(const v8bf*)(qrow + 16 + hi8);
      v8bf l1 = *(const v8bf*)(qrow + 32 + hi8);
      v8bf h1 = *(const v8bf*)(qrow + 48 + hi8);
#pragma unroll
      for (int e = 0; e < 8; ++e) {
        a0[e] = l0[e]; a0[e + 8] = h0[e];
        a1[e] = l1[e]; a1[e + 8] = h1[e];
      }
    }
    const __bf16* ktb = kt + (size_t)bh * D_DIM * S_DIM;
    for (int s0 = wave * 16; s0 < S_DIM; s0 += 128) {
      // B fragment (32x16 bf16): N = lane&15 (s index), K = 16*(e>>3) + 8*(lane>>4) + (e&7) (d index)
      v16bf b0, b1;
#pragma unroll
      for (int e = 0; e < 16; ++e) {
        const int kk = 16 * (e >> 3) + hi8 + (e & 7);
        b0[e] = ktb[(size_t)kk        * S_DIM + s0 + m16];
        b1[e] = ktb[(size_t)(kk + 32) * S_DIM + s0 + m16];
      }
      v8f c = {};
      c = __builtin_amdgcn_wmma_f32_16x16x32_bf16(false, a0, false, b0, (short)0, c, false, false);
      c = __builtin_amdgcn_wmma_f32_16x16x32_bf16(false, a1, false, b1, (short)0, c, false, false);
      const int n     = lane & 15;
      const int mbase = (lane >> 4) * 8;      // C/D layout: VGPR r -> M = mbase+r
#pragma unroll
      for (int r = 0; r < 8; ++r)
        sc[(mbase + r) * S_DIM + s0 + n] = c[r];
    }
    __syncthreads();

    // ---- Pass 2: row softmax (half-wave per row, shuffle reductions) ----
    {
      const int row = tid >> 4;               // == wave*2 + (lane>>4)
      const int c16 = tid & 15;
      float pmax = -3.4e38f;
      for (int s = c16; s < S_DIM; s += 16)
        pmax = fmaxf(pmax, sc[row * S_DIM + s] * scale);
#pragma unroll
      for (int off = 8; off > 0; off >>= 1)
        pmax = fmaxf(pmax, __shfl_xor(pmax, off, 16));
      float psum = 0.0f;
      for (int s = c16; s < S_DIM; s += 16) {
        float p = __expf(sc[row * S_DIM + s] * scale - pmax);
        sc[row * S_DIM + s] = p;              // unnormalized exp
        psum += p;
      }
#pragma unroll
      for (int off = 8; off > 0; off >>= 1)
        psum += __shfl_xor(psum, off, 16);
      const float inv = 1.0f / psum;
      for (int s = c16; s < S_DIM; s += 16) {
        float p = sc[row * S_DIM + s] * inv;
        pb[row * S_DIM + s] = (__bf16)p;      // packed bf16 P for pass 3
        avg[row * S_DIM + s] += p * (1.0f / (float)H_DIM);
      }
    }
    __syncthreads();

    // ---- Pass 3: ctx = P V (waves 0..3, each owns one 16-wide d chunk) ----
    if (wave < 4) {
      const int dcol = wave * 16;
      const __bf16* vbh = vb + (size_t)bh * S_DIM * D_DIM;
      const __bf16* prow = pb + (size_t)m16 * S_DIM;
      v8f c = {};
      for (int sch = 0; sch < S_DIM; sch += 32) {
        // A fragment straight from packed-bf16 LDS: two 16B ds loads
        v8bf lo = *(const v8bf*)(prow + sch + hi8);
        v8bf hi = *(const v8bf*)(prow + sch + 16 + hi8);
        v16bf ap;
#pragma unroll
        for (int e = 0; e < 8; ++e) { ap[e] = lo[e]; ap[e + 8] = hi[e]; }
        // B fragment: N = lane&15 (d index), K = 16*(e>>3) + 8*(lane>>4) + (e&7) (s index)
        v16bf bv;
#pragma unroll
        for (int e = 0; e < 16; ++e)
          bv[e] = vbh[(size_t)(sch + 16 * (e >> 3) + hi8 + (e & 7)) * D_DIM + dcol + m16];
        c = __builtin_amdgcn_wmma_f32_16x16x32_bf16(false, ap, false, bv, (short)0, c, false, false);
      }
      const int n     = lane & 15;
      const int mbase = (lane >> 4) * 8;
#pragma unroll
      for (int r = 0; r < 8; ++r) {
        int row = t0 + mbase + r;
        ctx[((size_t)row * B_DIM + b) * E_DIM + h * D_DIM + dcol + n] = (__bf16)c[r];
      }
    }
    __syncthreads();
  }

  // attn_avg out: (B,T,S)
  for (int i = tid; i < 16 * S_DIM; i += 256) {
    int row = i >> 11;
    int s   = i & (S_DIM - 1);
    attn_out[((size_t)b * T_DIM + (t0 + row)) * S_DIM + s] = avg[i];
  }
}

// ---------------------------------------------------------------------------
// Fused-heads projection: out = ctx @ W^T + bias + residual(query)
// M = T*B = 4096 rows, N = K = 1024.  4 waves/block, 16x64 output tile.
// ---------------------------------------------------------------------------
__global__ __launch_bounds__(128)
void proj_kernel(const __bf16* __restrict__ ctx, const __bf16* __restrict__ Wt,
                 const float* __restrict__ bias, const float* __restrict__ resid,
                 float* __restrict__ out) {
  const int tid  = threadIdx.x;
  const int lane = tid & 31;
  const int wave = tid >> 5;            // 0..3
  const int hi8  = (lane >> 4) * 8;
  const int m16  = lane & 15;

  const int r0 = blockIdx.x * 16;                 // row tile in M
  const int n0 = blockIdx.y * 64 + wave * 16;     // column tile in N

  const __bf16* arow = ctx + (size_t)(r0 + m16) * E_DIM;
  v8f c = {};
  for (int kc = 0; kc < E_DIM; kc += 32) {
    v16bf a;
    v8bf lo = *(const v8bf*)(arow + kc + hi8);
    v8bf hi = *(const v8bf*)(arow + kc + 16 + hi8);
#pragma unroll
    for (int e = 0; e < 8; ++e) { a[e] = lo[e]; a[e + 8] = hi[e]; }
    v16bf bw;
#pragma unroll
    for (int e = 0; e < 16; ++e)
      bw[e] = Wt[(size_t)(kc + 16 * (e >> 3) + hi8 + (e & 7)) * E_DIM + n0 + m16];
    c = __builtin_amdgcn_wmma_f32_16x16x32_bf16(false, a, false, bw, (short)0, c, false, false);
  }
  const int n     = lane & 15;
  const int mbase = (lane >> 4) * 8;
#pragma unroll
  for (int r = 0; r < 8; ++r) {
    size_t row = r0 + mbase + r;
    int    col = n0 + n;
    out[row * E_DIM + col] = c[r] + bias[col] + resid[row * E_DIM + col];
  }
}

// ---------------------------------------------------------------------------
extern "C" void kernel_launch(void* const* d_in, const int* in_sizes, int n_in,
                              void* d_out, int out_size, void* d_ws, size_t ws_size,
                              hipStream_t stream) {
  const float* q    = (const float*)d_in[0];
  const float* k    = (const float*)d_in[1];
  const float* v    = (const float*)d_in[2];
  const float* qp   = (const float*)d_in[3];
  const float* kp   = (const float*)d_in[4];
  const float* W    = (const float*)d_in[5];
  const float* bias = (const float*)d_in[6];

  float* out  = (float*)d_out;
  float* attn = out + (size_t)T_DIM * B_DIM * E_DIM;

  // Workspace layout (34 MB total):
  char*   ws  = (char*)d_ws;
  __bf16* qb  = (__bf16*)(ws);                        // 8 MB [B][H][T][D]
  __bf16* kt  = (__bf16*)(ws + (size_t)(8u  << 20));  // 8 MB [B][H][D][S]
  __bf16* vb  = (__bf16*)(ws + (size_t)(16u << 20));  // 8 MB [B][H][S][D]
  __bf16* Wt  = (__bf16*)(ws + (size_t)(24u << 20));  // 2 MB [K][N]
  __bf16* ctx = (__bf16*)(ws + (size_t)(26u << 20));  // 8 MB [T*B][E]

  const int nQKV = B_DIM * H_DIM * T_DIM * D_DIM;     // 4,194,304
  prep_qkv<<<nQKV / 256, 256, 0, stream>>>(q, qp, k, kp, v, qb, kt, vb);
  prep_w<<<(E_DIM * E_DIM) / 256, 256, 0, stream>>>(W, Wt);

  // sc (128KB) + avg (128KB) + pb (64KB) == 320 KB (WGP LDS max)
  size_t shm = (size_t)(16 * S_DIM) * (4 + 4 + 2);
  attn_kernel<<<dim3(T_DIM / 16, B_DIM), 256, shm, stream>>>(qb, kt, vb, ctx, attn);

  proj_kernel<<<dim3((T_DIM * B_DIM) / 16, E_DIM / 64), 128, 0, stream>>>(
      ctx, Wt, bias, q, out);
}
